// PointNet_plusplus_28157805593264
// MI455X (gfx1250) — compile-verified, matrix operand order checked by reading, not run on hardware
//
#include <hip/hip_runtime.h>
#include <math.h>

typedef unsigned short u16;
typedef unsigned int   u32;
typedef __attribute__((ext_vector_type(16))) __bf16 v16bf;
typedef __attribute__((ext_vector_type(8)))  float  v8f;

// one WMMA 16-bit operand fragment = 16 bf16 per lane = two contiguous 16B runs
union Frag { uint4 q[2]; v16bf v; u16 s[16]; };

__device__ __forceinline__ u16 f2bf(float f) {
  u32 u = __float_as_uint(f);
  u32 r = u + 0x7FFFu + ((u >> 16) & 1u);   // round-to-nearest-even
  return (u16)(r >> 16);
}
__device__ __forceinline__ float bf2f(u16 h) {
  return __uint_as_float(((u32)h) << 16);
}
__device__ __forceinline__ float sigm(float x) { return 1.0f / (1.0f + __expf(-x)); }

// ---------------------------------------------------------------------------
// bf16 WMMA GEMM:  C[M,N] = A[M,K] @ W (+bias[N]) (+addC[M,N])
//   A : bf16 row-major, row stride K, K % 32 == 0 (callers zero-pad)
//   WT: bf16 row-major (N rows, K cols) == W transposed, K % 32 == 0
// Block tile 64x64, 8 waves; wave = one 16-row M tile x two 16-col N tiles
// -> 2x v_wmma_f32_16x16x32_bf16 per K-step sharing one A fragment.
// Edge rows are handled by CLAMPING the tile row index (duplicate valid data,
// never NaN) so staging stays a single unconditional global_load_b128 +
// ds_store_b128 per thread per tile; results are masked at the C store.
// ---------------------------------------------------------------------------
#define TBM 64
#define TBN 64
#define TBK 32

__global__ __launch_bounds__(256) void k_gemm(
    const u16* __restrict__ A, const u16* __restrict__ WT,
    const float* __restrict__ bias, const float* __restrict__ addC,
    float* __restrict__ C, int M, int N, int K) {
  __shared__ __align__(16) u16 sA[TBM * TBK];   // [m][k]
  __shared__ __align__(16) u16 sB[TBN * TBK];   // [n][k]
  const int tid  = threadIdx.x;
  const int lane = tid & 31;
  const int wave = tid >> 5;
  const int wm   = (wave >> 1) << 4;            // 0,16,32,48
  const int wn   = (wave & 1) << 5;             // 0,32  (two 16-wide N tiles)
  const int bm   = blockIdx.y * TBM;
  const int bn   = blockIdx.x * TBN;
  // staging assignment: each thread owns one 8-elem (16B) chunk of each tile
  const int sr = tid >> 2;                      // 0..63 tile row
  const int sc = (tid & 3) << 3;                // 0,8,16,24 k-offset
  const int gm = min(bm + sr, M - 1);           // clamped tile rows
  const int gn = min(bn + sr, N - 1);
  const u16* __restrict__ pA = A  + (size_t)gm * K + sc;
  const u16* __restrict__ pB = WT + (size_t)gn * K + sc;

  v8f acc0 = {0.f, 0.f, 0.f, 0.f, 0.f, 0.f, 0.f, 0.f};
  v8f acc1 = acc0;

  const int arow = wm + (lane & 15);
  const int b0row = wn + (lane & 15);
  const int base = (lane & 16) >> 1;            // 0 or 8

  for (int k0 = 0; k0 < K; k0 += TBK) {
    *(uint4*)(&sA[sr * TBK + sc]) = *(const uint4*)(pA + k0);
    *(uint4*)(&sB[sr * TBK + sc]) = *(const uint4*)(pB + k0);
    __syncthreads();
    if (k0 + TBK < K)                           // prefetch next A chunk toward near cache
      __builtin_prefetch(pA + k0 + TBK, 0, 3);
    // fragments: elements e=0..7 -> k = base+e ; e=8..15 -> k = base+16+e-8
    Frag fa, fb0, fb1;
    {
      const uint4* pa = (const uint4*)(&sA[arow * TBK + base]);
      fa.q[0] = pa[0]; fa.q[1] = pa[2];
      const uint4* pb0 = (const uint4*)(&sB[b0row * TBK + base]);
      fb0.q[0] = pb0[0]; fb0.q[1] = pb0[2];
      const uint4* pb1 = (const uint4*)(&sB[(b0row + 16) * TBK + base]);
      fb1.q[0] = pb1[0]; fb1.q[1] = pb1[2];
    }
    acc0 = __builtin_amdgcn_wmma_f32_16x16x32_bf16(false, fa.v, false, fb0.v,
                                                   (short)0, acc0, false, false);
    acc1 = __builtin_amdgcn_wmma_f32_16x16x32_bf16(false, fa.v, false, fb1.v,
                                                   (short)0, acc1, false, false);
    __syncthreads();
  }
  // C layout per tile: N = lane&15, M = vgpr + (lane>=16)*8
  const int n0 = bn + wn + (lane & 15);
  const int n1 = n0 + 16;
  const int mb = bm + wm + ((lane >> 4) << 3);
#pragma unroll
  for (int v = 0; v < 8; ++v) {
    int m = mb + v;
    if (m < M) {
      if (n0 < N) {
        float val = acc0[v] + (bias ? bias[n0] : 0.0f);
        if (addC) val += addC[(size_t)m * N + n0];
        C[(size_t)m * N + n0] = val;
      }
      if (n1 < N) {
        float val = acc1[v] + (bias ? bias[n1] : 0.0f);
        if (addC) val += addC[(size_t)m * N + n1];
        C[(size_t)m * N + n1] = val;
      }
    }
  }
}

// ---------------------------------------------------------------------------
// Small utility kernels
// ---------------------------------------------------------------------------
__global__ void k_fill_f32(float* p, float v, long long n) {
  long long i = (long long)blockIdx.x * blockDim.x + threadIdx.x;
  if (i < n) p[i] = v;
}
__global__ void k_fill_u16(u16* p, int v, long long n) {
  long long i = (long long)blockIdx.x * blockDim.x + threadIdx.x;
  if (i < n) p[i] = (u16)v;
}
__global__ void k_convert_bf16(const float* __restrict__ in, u16* __restrict__ out, long long n) {
  long long i = (long long)blockIdx.x * blockDim.x + threadIdx.x;
  if (i < n) out[i] = f2bf(in[i]);
}
// W (K,N) row-major f32  ->  WT (N, Kp) row-major bf16, zero pad k in [K,Kp)
__global__ void k_convert_wt(const float* __restrict__ in, u16* __restrict__ out,
                             int K, int N, int Kp) {
  long long i = (long long)blockIdx.x * blockDim.x + threadIdx.x;
  if (i >= (long long)N * Kp) return;
  int n = (int)(i / Kp), k = (int)(i % Kp);
  out[i] = (k < K) ? f2bf(in[(size_t)k * N + n]) : (u16)0;
}
__global__ void k_add_vec(const float* __restrict__ a, const float* __restrict__ b,
                          float* __restrict__ o, long long n) {
  long long i = (long long)blockIdx.x * blockDim.x + threadIdx.x;
  if (i < n) o[i] = a[i] + b[i];
}
__global__ void k_transpose_x(const float* __restrict__ x, float* __restrict__ pts,
                              float* __restrict__ xyz, int Bn, int N) {
  long long i = (long long)blockIdx.x * blockDim.x + threadIdx.x;
  if (i >= (long long)Bn * N) return;
  int b = (int)(i / N), n = (int)(i % N);
#pragma unroll
  for (int c = 0; c < 4; ++c) {
    float v = x[((size_t)b * 4 + c) * N + n];
    pts[(size_t)i * 4 + c] = v;
    if (c < 3) xyz[(size_t)i * 3 + c] = v;
  }
}

// ---------------------------------------------------------------------------
// Farthest point sampling: one block per batch; dist[] kept in LDS.
// ---------------------------------------------------------------------------
__global__ __launch_bounds__(256) void k_fps(const float* __restrict__ xyz, int Npts, int np,
                                             int* __restrict__ out_idx) {
  __shared__ float dist[8192];
  __shared__ float rv[256];
  __shared__ int   ri[256];
  __shared__ int   s_far;
  const int b = blockIdx.x, tid = threadIdx.x;
  const float* xb = xyz + (size_t)b * Npts * 3;
  for (int i = tid; i < Npts; i += 256) dist[i] = 1e10f;
  if (tid == 0) s_far = 0;
  __syncthreads();
  for (int it = 0; it < np; ++it) {
    int far = s_far;
    float cx = xb[far * 3], cy = xb[far * 3 + 1], cz = xb[far * 3 + 2];
    float bv = -1.0f; int bi = 0;
    for (int i = tid; i < Npts; i += 256) {
      float dx = xb[i * 3] - cx, dy = xb[i * 3 + 1] - cy, dz = xb[i * 3 + 2] - cz;
      float d = dx * dx + dy * dy + dz * dz;
      float dd = fminf(dist[i], d);
      dist[i] = dd;
      if (dd > bv) { bv = dd; bi = i; }
    }
    rv[tid] = bv; ri[tid] = bi;
    __syncthreads();
    for (int s = 128; s > 0; s >>= 1) {
      if (tid < s && rv[tid + s] > rv[tid]) { rv[tid] = rv[tid + s]; ri[tid] = ri[tid + s]; }
      __syncthreads();
    }
    if (tid == 0) { out_idx[(size_t)b * np + it] = far; s_far = ri[0]; }
    __syncthreads();
  }
}

__global__ void k_gather_xyz(const float* __restrict__ xyz, const int* __restrict__ fidx,
                             float* __restrict__ nxyz, int Npts, int S, long long total) {
  long long i = (long long)blockIdx.x * blockDim.x + threadIdx.x;
  if (i >= total) return;
  int b = (int)(i / S);
  int p = fidx[i];
#pragma unroll
  for (int k = 0; k < 3; ++k)
    nxyz[(size_t)i * 3 + k] = xyz[((size_t)b * Npts + p) * 3 + k];
}

__global__ void k_ball_query(const float* __restrict__ xyz, const float* __restrict__ nxyz,
                             int* __restrict__ gidx, int Npts, int S, int ns, float r2,
                             long long total) {
  long long i = (long long)blockIdx.x * blockDim.x + threadIdx.x;
  if (i >= total) return;
  int b = (int)(i / S);
  float qx = nxyz[(size_t)i * 3], qy = nxyz[(size_t)i * 3 + 1], qz = nxyz[(size_t)i * 3 + 2];
  const float* xb = xyz + (size_t)b * Npts * 3;
  int cnt = 0, first = -1;
  for (int j = 0; j < Npts && cnt < ns; ++j) {
    float dx = xb[j * 3] - qx, dy = xb[j * 3 + 1] - qy, dz = xb[j * 3 + 2] - qz;
    if (dx * dx + dy * dy + dz * dz <= r2) {
      if (first < 0) first = j;
      gidx[(size_t)i * ns + cnt] = j;
      ++cnt;
    }
  }
  if (first < 0) first = 0;
  for (; cnt < ns; ++cnt) gidx[(size_t)i * ns + cnt] = first;
}

// Grouped A rows (bf16, stride Cp): [xyz[idx]-new_xyz (3) | feats[idx] (Cf) | 0 pad]
__global__ void k_group(const float* __restrict__ xyz, const float* __restrict__ nxyz,
                        const float* __restrict__ feats, int Cf, const int* __restrict__ gidx,
                        u16* __restrict__ A, int Npts, int S, int ns, int Cp, long long total) {
  long long i = (long long)blockIdx.x * blockDim.x + threadIdx.x;
  if (i >= total) return;
  int s = (int)((i / ns) % S);
  int b = (int)(i / ((long long)ns * S));
  int p = gidx[i];
  size_t base = (size_t)i * Cp;
#pragma unroll
  for (int k = 0; k < 3; ++k)
    A[base + k] = f2bf(xyz[((size_t)b * Npts + p) * 3 + k] - nxyz[((size_t)b * S + s) * 3 + k]);
  for (int c = 0; c < Cf; ++c)
    A[base + 3 + c] = f2bf(feats[((size_t)b * Npts + p) * Cf + c]);
  for (int c = 3 + Cf; c < Cp; ++c)
    A[base + c] = (u16)0;
}

// Per-channel mean/var over M rows (population variance).
__global__ __launch_bounds__(256) void k_meanvar(const float* __restrict__ Z, float* __restrict__ mean,
                                                 float* __restrict__ var, int M, int C) {
  __shared__ float s1[256], s2[256];
  int c = blockIdx.x, tid = threadIdx.x;
  float a = 0.f, b = 0.f;
  for (int r = tid; r < M; r += 256) {
    float v = Z[(size_t)r * C + c];
    a += v; b += v * v;
  }
  s1[tid] = a; s2[tid] = b;
  __syncthreads();
  for (int s = 128; s > 0; s >>= 1) {
    if (tid < s) { s1[tid] += s1[tid + s]; s2[tid] += s2[tid + s]; }
    __syncthreads();
  }
  if (tid == 0) {
    float m = s1[0] / (float)M;
    mean[c] = m;
    var[c] = s2[0] / (float)M - m * m;
  }
}

// BN + activation (0=none,1=relu,2=leaky0.2); optional f32 and bf16 outputs.
__global__ void k_bn_act(const float* __restrict__ Z, const float* __restrict__ mean,
                         const float* __restrict__ var, const float* __restrict__ gamma,
                         const float* __restrict__ beta, float* __restrict__ f32out,
                         u16* __restrict__ bfout, long long total, int C, int act) {
  long long i = (long long)blockIdx.x * blockDim.x + threadIdx.x;
  if (i >= total) return;
  int c = (int)(i % C);
  float y = (Z[i] - mean[c]) * rsqrtf(var[c] + 1e-5f) * gamma[c] + beta[c];
  if (act == 1) y = fmaxf(y, 0.f);
  else if (act == 2) y = (y > 0.f) ? y : 0.2f * y;
  if (f32out) f32out[i] = y;
  if (bfout)  bfout[i]  = f2bf(y);
}

// Max over nsample group members (reads bf16 activations).
__global__ void k_maxpool(const u16* __restrict__ act, float* __restrict__ out,
                          int ns, int C, long long total) {
  long long i = (long long)blockIdx.x * blockDim.x + threadIdx.x;
  if (i >= total) return;
  long long r = i / C; int c = (int)(i % C);
  float m = -1e30f;
  for (int j = 0; j < ns; ++j)
    m = fmaxf(m, bf2f(act[((size_t)r * ns + j) * C + c]));
  out[i] = m;
}

// im2col from (B, Cin, L) signal.
__global__ void k_im2col_sig(const float* __restrict__ sig, u16* __restrict__ A,
                             int Cin, int Lin, int Lout, int ks, int stride, int pad,
                             long long total) {
  long long i = (long long)blockIdx.x * blockDim.x + threadIdx.x;
  if (i >= total) return;
  int c = (int)(i % Cin);
  long long t1 = i / Cin;
  int kk = (int)(t1 % ks);
  long long t2 = t1 / ks;
  int t = (int)(t2 % Lout);
  int b = (int)(t2 / Lout);
  int pos = t * stride - pad + kk;
  float v = (pos >= 0 && pos < Lin) ? sig[((size_t)b * Cin + c) * Lin + pos] : 0.0f;
  A[i] = f2bf(v);
}

// im2col from row-major activations ((b*Lin+pos), Cin).
__global__ void k_im2col_act(const float* __restrict__ act, u16* __restrict__ A,
                             int Cin, int Lin, int Lout, int ks, int stride, int pad,
                             long long total) {
  long long i = (long long)blockIdx.x * blockDim.x + threadIdx.x;
  if (i >= total) return;
  int c = (int)(i % Cin);
  long long t1 = i / Cin;
  int kk = (int)(t1 % ks);
  long long t2 = t1 / ks;
  int t = (int)(t2 % Lout);
  int b = (int)(t2 / Lout);
  int pos = t * stride - pad + kk;
  float v = (pos >= 0 && pos < Lin) ? act[((size_t)b * Lin + pos) * Cin + c] : 0.0f;
  A[i] = f2bf(v);
}

// (b,t,c) activations -> (t*B+b, c) bf16 sequence rows.
__global__ void k_pack_seq(const float* __restrict__ act, u16* __restrict__ A,
                           int Bn, int T, int C, long long total) {
  long long i = (long long)blockIdx.x * blockDim.x + threadIdx.x;
  if (i >= total) return;
  int c = (int)(i % C);
  int b = (int)((i / C) % Bn);
  int t = (int)(i / ((long long)C * Bn));
  A[i] = f2bf(act[((size_t)b * T + t) * C + c]);
}

// LSTM pointwise gate math; h written to bf16 state and packed hs buffer.
__global__ void k_lstm_pointwise(const float* __restrict__ G, float* __restrict__ cst,
                                 u16* __restrict__ hbf, u16* __restrict__ hs,
                                 int t_store, int col0, int Bn, int H) {
  int i = blockIdx.x * blockDim.x + threadIdx.x;
  if (i >= Bn * H) return;
  int b = i / H, j = i % H;
  const float* g = G + (size_t)b * 4 * H;
  float ig = sigm(g[j]);
  float fg = sigm(g[H + j]);
  float gg = tanhf(g[2 * H + j]);
  float og = sigm(g[3 * H + j]);
  float c = fg * cst[i] + ig * gg;
  float h = og * tanhf(c);
  cst[i] = c;
  u16 hb = f2bf(h);
  hbf[i] = hb;
  hs[((size_t)t_store * Bn + b) * (2 * H) + col0 + j] = hb;
}

// feats = max over time of out[(t*B+b), c], c < 128.
__global__ void k_ecg_max(const float* __restrict__ Z, float* __restrict__ ecg, int T, int Bn) {
  int i = blockIdx.x * blockDim.x + threadIdx.x;
  if (i >= Bn * 128) return;
  int b = i / 128, c = i % 128;
  float m = -1e30f;
  for (int t = 0; t < T; ++t)
    m = fmaxf(m, Z[((size_t)t * Bn + b) * 256 + c]);
  ecg[i] = m;
}

// feat = concat(l3, broadcast ecg) over channel dim.
__global__ void k_concat_feat(const float* __restrict__ l3, const float* __restrict__ ecg,
                              float* __restrict__ feat, int S, int C3, int CE, long long total) {
  long long i = (long long)blockIdx.x * blockDim.x + threadIdx.x;
  if (i >= total) return;
  int C = C3 + CE;
  int c = (int)(i % C);
  int s = (int)((i / C) % S);
  int b = (int)(i / ((long long)C * S));
  feat[i] = (c < C3) ? l3[((size_t)b * S + s) * C3 + c] : ecg[(size_t)b * CE + (c - C3)];
}

// 3-NN inverse-distance interpolation + channel concat -> bf16 A rows (stride Cp).
__global__ void k_interp_concat(const float* __restrict__ xyz1, const float* __restrict__ xyz2,
                                const float* __restrict__ p1, int C1,
                                const float* __restrict__ p2, int C2,
                                u16* __restrict__ A, int N1, int N2, int Cp, long long total) {
  long long i = (long long)blockIdx.x * blockDim.x + threadIdx.x;
  if (i >= total) return;
  int b = (int)(i / N1);
  float px = xyz1[(size_t)i * 3], py = xyz1[(size_t)i * 3 + 1], pz = xyz1[(size_t)i * 3 + 2];
  const float* xb = xyz2 + (size_t)b * N2 * 3;
  float d0 = 1e30f, d1 = 1e30f, d2 = 1e30f;
  int i0 = 0, i1 = 0, i2 = 0;
  for (int j = 0; j < N2; ++j) {
    float dx = xb[j * 3] - px, dy = xb[j * 3 + 1] - py, dz = xb[j * 3 + 2] - pz;
    float d = dx * dx + dy * dy + dz * dz;
    if (d < d0)      { d2 = d1; i2 = i1; d1 = d0; i1 = i0; d0 = d; i0 = j; }
    else if (d < d1) { d2 = d1; i2 = i1; d1 = d;  i1 = j; }
    else if (d < d2) { d2 = d;  i2 = j; }
  }
  float w0 = 1.f / (d0 + 1e-8f), w1 = 1.f / (d1 + 1e-8f), w2 = 1.f / (d2 + 1e-8f);
  float ws = w0 + w1 + w2;
  w0 /= ws; w1 /= ws; w2 /= ws;
  int C = C1 + C2;
  size_t base = (size_t)i * Cp;
  for (int c = 0; c < C1; ++c)
    A[base + c] = f2bf(p1[(size_t)i * C1 + c]);
  const float* q0 = p2 + ((size_t)b * N2 + i0) * C2;
  const float* q1 = p2 + ((size_t)b * N2 + i1) * C2;
  const float* q2 = p2 + ((size_t)b * N2 + i2) * C2;
  for (int c = 0; c < C2; ++c)
    A[base + C1 + c] = f2bf(w0 * q0[c] + w1 * q1[c] + w2 * q2[c]);
  for (int c = C; c < Cp; ++c)
    A[base + c] = (u16)0;
}

// ---------------------------------------------------------------------------
// Host orchestration
// ---------------------------------------------------------------------------
extern "C" void kernel_launch(void* const* d_in, const int* in_sizes, int n_in,
                              void* d_out, int out_size, void* d_ws, size_t ws_size,
                              hipStream_t stream) {
  (void)in_sizes; (void)n_in; (void)out_size; (void)ws_size;

  char* wbase = (char*)d_ws;
  size_t cur = 0;
  auto alloc = [&](size_t bytes) -> void* {
    size_t a = (cur + 255) & ~(size_t)255;
    cur = a + bytes;
    return (void*)(wbase + a);
  };
  auto pad32 = [](int k) { return (k + 31) & ~31; };

  // Model constants
  const int Bb = 16, N0 = 8192;
  const int S1 = 512, NS1 = 64, S2 = 128, NS2 = 64, S3 = 16, NS3 = 32;
  const int L = 2048, CIN = 8, C1 = 128, C2 = 256;
  const int L1 = 1019, L2 = 504, T = 504, H = 512, G4 = 2048;

  // Workspace arena
  u16*   Abuf   = (u16*)  alloc((size_t)33554432 * sizeof(u16));    // bf16 A ping (padded strides)
  float* Zbuf   = (float*)alloc((size_t)67108864 * sizeof(float));  // f32 GEMM out
  u16*   Wbf    = (u16*)  alloc((size_t)1048576 * sizeof(u16));     // bf16 W^T per layer
  float* meanb  = (float*)alloc(1024 * sizeof(float));
  float* varb   = (float*)alloc(1024 * sizeof(float));
  float* cact1  = (float*)alloc((size_t)Bb * L1 * C1 * sizeof(float));
  float* cact2  = (float*)alloc((size_t)Bb * L2 * C2 * sizeof(float));
  float* xpf    = (float*)alloc((size_t)T * Bb * G4 * sizeof(float));
  float* xpb    = (float*)alloc((size_t)T * Bb * G4 * sizeof(float));
  u16*   WihBf  = (u16*)  alloc((size_t)G4 * 256 * sizeof(u16));    // (N=2048, K=256) = Wih as-is
  u16*   WihBb  = (u16*)  alloc((size_t)G4 * 256 * sizeof(u16));
  u16*   WhhBf  = (u16*)  alloc((size_t)G4 * H * sizeof(u16));      // (N=2048, K=512) = Whh as-is
  u16*   WhhBb  = (u16*)  alloc((size_t)G4 * H * sizeof(u16));
  float* biasF  = (float*)alloc(G4 * sizeof(float));
  float* biasB  = (float*)alloc(G4 * sizeof(float));
  u16*   hbF    = (u16*)  alloc((size_t)Bb * H * sizeof(u16));
  u16*   hbB    = (u16*)  alloc((size_t)Bb * H * sizeof(u16));
  float* cF     = (float*)alloc((size_t)Bb * H * sizeof(float));
  float* cB     = (float*)alloc((size_t)Bb * H * sizeof(float));
  float* Gf     = (float*)alloc((size_t)Bb * G4 * sizeof(float));
  float* Gb     = (float*)alloc((size_t)Bb * G4 * sizeof(float));
  u16*   hsbuf  = (u16*)  alloc((size_t)T * Bb * 1024 * sizeof(u16));
  float* ecg    = (float*)alloc((size_t)Bb * 128 * sizeof(float));
  float* pts    = (float*)alloc((size_t)Bb * N0 * 4 * sizeof(float));
  float* xyz    = (float*)alloc((size_t)Bb * N0 * 3 * sizeof(float));
  int*   idx1   = (int*)  alloc((size_t)Bb * S1 * sizeof(int));
  float* nxyz1  = (float*)alloc((size_t)Bb * S1 * 3 * sizeof(float));
  int*   gidx1  = (int*)  alloc((size_t)Bb * S1 * NS1 * sizeof(int));
  int*   idx2   = (int*)  alloc((size_t)Bb * S2 * sizeof(int));
  float* nxyz2  = (float*)alloc((size_t)Bb * S2 * 3 * sizeof(float));
  int*   gidx2  = (int*)  alloc((size_t)Bb * S2 * NS2 * sizeof(int));
  int*   idx3   = (int*)  alloc((size_t)Bb * S3 * sizeof(int));
  float* nxyz3  = (float*)alloc((size_t)Bb * S3 * 3 * sizeof(float));
  int*   gidx3  = (int*)  alloc((size_t)Bb * S3 * NS3 * sizeof(int));
  float* l1f    = (float*)alloc((size_t)Bb * S1 * 128 * sizeof(float));
  float* l2f    = (float*)alloc((size_t)Bb * S2 * 256 * sizeof(float));
  float* l3f    = (float*)alloc((size_t)Bb * S3 * 1024 * sizeof(float));
  float* featf  = (float*)alloc((size_t)Bb * S3 * 1152 * sizeof(float));
  float* l2n    = (float*)alloc((size_t)Bb * S2 * 256 * sizeof(float));
  float* l1n    = (float*)alloc((size_t)Bb * S1 * 128 * sizeof(float));

  auto F = [&](int i) { return (const float*)d_in[i]; };
  auto grid1 = [&](long long n) { return dim3((unsigned)((n + 255) / 256), 1, 1); };
  auto gemm = [&](const u16* A, const u16* WT, const float* bias, const float* addC,
                  float* C, int M, int N, int K) {
    dim3 g((N + TBN - 1) / TBN, (M + TBM - 1) / TBM);
    k_gemm<<<g, 256, 0, stream>>>(A, WT, bias, addC, C, M, N, K);
  };
  auto conv_wt = [&](int wi, int K, int N, int Kp) {
    k_convert_wt<<<grid1((long long)N * Kp), 256, 0, stream>>>(F(wi), Wbf, K, N, Kp);
  };
  // one MLP layer: A(bf16,stride Kp) @ W -> BN -> act -> {f32out?, bf16out?}
  auto mlp_layer = [&](int wi, int M, int K, int Nc, int act, float* f32out, u16* bfout) {
    int Kp = pad32(K);
    conv_wt(wi, K, Nc, Kp);
    gemm(Abuf, Wbf, F(wi + 1), nullptr, Zbuf, M, Nc, Kp);
    k_meanvar<<<Nc, 256, 0, stream>>>(Zbuf, meanb, varb, M, Nc);
    k_bn_act<<<grid1((long long)M * Nc), 256, 0, stream>>>(
        Zbuf, meanb, varb, F(wi + 2), F(wi + 3), f32out, bfout, (long long)M * Nc, Nc, act);
  };

  // ============================ CRNN ============================
  // conv1: im2col(signal) -> GEMM (K=16*8=128, N=128) -> BN+leaky
  k_im2col_sig<<<grid1((long long)Bb * L1 * 16 * CIN), 256, 0, stream>>>(
      F(1), Abuf, CIN, L, L1, 16, 2, 2, (long long)Bb * L1 * 16 * CIN);
  conv_wt(72, 128, 128, 128);
  gemm(Abuf, Wbf, F(73), nullptr, Zbuf, Bb * L1, 128, 128);
  k_meanvar<<<128, 256, 0, stream>>>(Zbuf, meanb, varb, Bb * L1, 128);
  k_bn_act<<<grid1((long long)Bb * L1 * 128), 256, 0, stream>>>(
      Zbuf, meanb, varb, F(74), F(75), cact1, (u16*)nullptr, (long long)Bb * L1 * 128, 128, 2);
  // conv2: im2col(cact1) -> GEMM (K=16*128=2048, N=256) -> BN+leaky
  k_im2col_act<<<grid1((long long)Bb * L2 * 16 * C1), 256, 0, stream>>>(
      cact1, Abuf, C1, L1, L2, 16, 2, 2, (long long)Bb * L2 * 16 * C1);
  conv_wt(76, 2048, 256, 2048);
  gemm(Abuf, Wbf, F(77), nullptr, Zbuf, Bb * L2, 256, 2048);
  k_meanvar<<<256, 256, 0, stream>>>(Zbuf, meanb, varb, Bb * L2, 256);
  k_bn_act<<<grid1((long long)Bb * L2 * 256), 256, 0, stream>>>(
      Zbuf, meanb, varb, F(78), F(79), cact2, (u16*)nullptr, (long long)Bb * L2 * 256, 256, 2);
  // sequence (T, B, 256) bf16
  k_pack_seq<<<grid1((long long)T * Bb * 256), 256, 0, stream>>>(
      cact2, Abuf, Bb, T, 256, (long long)T * Bb * 256);
  // LSTM weights are already W^T layout (rows = 4H output cols, cols = input K)
  k_convert_bf16<<<grid1((long long)G4 * 256), 256, 0, stream>>>(F(80), WihBf, (long long)G4 * 256);
  k_convert_bf16<<<grid1((long long)G4 * H), 256, 0, stream>>>(F(81), WhhBf, (long long)G4 * H);
  k_add_vec<<<grid1(G4), 256, 0, stream>>>(F(82), F(83), biasF, G4);
  k_convert_bf16<<<grid1((long long)G4 * 256), 256, 0, stream>>>(F(84), WihBb, (long long)G4 * 256);
  k_convert_bf16<<<grid1((long long)G4 * H), 256, 0, stream>>>(F(85), WhhBb, (long long)G4 * H);
  k_add_vec<<<grid1(G4), 256, 0, stream>>>(F(86), F(87), biasB, G4);
  // x projections for all timesteps (both directions)
  gemm(Abuf, WihBf, biasF, nullptr, xpf, T * Bb, G4, 256);
  gemm(Abuf, WihBb, biasB, nullptr, xpb, T * Bb, G4, 256);
  // init states
  k_fill_u16<<<grid1(Bb * H), 256, 0, stream>>>(hbF, 0, (long long)Bb * H);
  k_fill_u16<<<grid1(Bb * H), 256, 0, stream>>>(hbB, 0, (long long)Bb * H);
  k_fill_f32<<<grid1(Bb * H), 256, 0, stream>>>(cF, 0.f, (long long)Bb * H);
  k_fill_f32<<<grid1(Bb * H), 256, 0, stream>>>(cB, 0.f, (long long)Bb * H);
  // recurrent loop (fwd + bwd interleaved)
  for (int t = 0; t < T; ++t) {
    gemm(hbF, WhhBf, nullptr, xpf + (size_t)t * Bb * G4, Gf, Bb, G4, H);
    k_lstm_pointwise<<<grid1(Bb * H), 256, 0, stream>>>(Gf, cF, hbF, hsbuf, t, 0, Bb, H);
    int tb = T - 1 - t;
    gemm(hbB, WhhBb, nullptr, xpb + (size_t)tb * Bb * G4, Gb, Bb, G4, H);
    k_lstm_pointwise<<<grid1(Bb * H), 256, 0, stream>>>(Gb, cB, hbB, hsbuf, tb, H, Bb, H);
  }
  // embed: rec(8064,1024) @ eW(1024,256) + eb, then max over time, keep 128
  conv_wt(88, 1024, 256, 1024);
  gemm(hsbuf, Wbf, F(89), nullptr, Zbuf, T * Bb, 256, 1024);
  k_ecg_max<<<grid1(Bb * 128), 256, 0, stream>>>(Zbuf, ecg, T, Bb);

  // ========================= PointNet++ =========================
  k_transpose_x<<<grid1((long long)Bb * N0), 256, 0, stream>>>(F(0), pts, xyz, Bb, N0);

  // ---- SA1 ----
  k_fps<<<Bb, 256, 0, stream>>>(xyz, N0, S1, idx1);
  k_gather_xyz<<<grid1((long long)Bb * S1), 256, 0, stream>>>(xyz, idx1, nxyz1, N0, S1, (long long)Bb * S1);
  k_ball_query<<<grid1((long long)Bb * S1), 256, 0, stream>>>(xyz, nxyz1, gidx1, N0, S1, NS1, 0.04f, (long long)Bb * S1);
  k_group<<<grid1((long long)Bb * S1 * NS1), 256, 0, stream>>>(
      xyz, nxyz1, pts, 4, gidx1, Abuf, N0, S1, NS1, pad32(7), (long long)Bb * S1 * NS1);
  const int M1 = Bb * S1 * NS1;  // 524288
  mlp_layer(2, M1, 7, 64, 1, nullptr, Abuf);
  mlp_layer(6, M1, 64, 64, 1, nullptr, Abuf);
  mlp_layer(10, M1, 64, 128, 1, nullptr, Abuf);
  k_maxpool<<<grid1((long long)Bb * S1 * 128), 256, 0, stream>>>(Abuf, l1f, NS1, 128, (long long)Bb * S1 * 128);

  // ---- SA2 ----
  k_fps<<<Bb, 256, 0, stream>>>(nxyz1, S1, S2, idx2);
  k_gather_xyz<<<grid1((long long)Bb * S2), 256, 0, stream>>>(nxyz1, idx2, nxyz2, S1, S2, (long long)Bb * S2);
  k_ball_query<<<grid1((long long)Bb * S2), 256, 0, stream>>>(nxyz1, nxyz2, gidx2, S1, S2, NS2, 0.16f, (long long)Bb * S2);
  k_group<<<grid1((long long)Bb * S2 * NS2), 256, 0, stream>>>(
      nxyz1, nxyz2, l1f, 128, gidx2, Abuf, S1, S2, NS2, pad32(131), (long long)Bb * S2 * NS2);
  const int M2 = Bb * S2 * NS2;  // 131072
  mlp_layer(14, M2, 131, 128, 1, nullptr, Abuf);
  mlp_layer(18, M2, 128, 128, 1, nullptr, Abuf);
  mlp_layer(22, M2, 128, 256, 1, nullptr, Abuf);
  k_maxpool<<<grid1((long long)Bb * S2 * 256), 256, 0, stream>>>(Abuf, l2f, NS2, 256, (long long)Bb * S2 * 256);

  // ---- SA3 ----
  k_fps<<<Bb, 256, 0, stream>>>(nxyz2, S2, S3, idx3);
  k_gather_xyz<<<grid1((long long)Bb * S3), 256, 0, stream>>>(nxyz2, idx3, nxyz3, S2, S3, (long long)Bb * S3);
  k_ball_query<<<grid1((long long)Bb * S3), 256, 0, stream>>>(nxyz2, nxyz3, gidx3, S2, S3, NS3, 0.64f, (long long)Bb * S3);
  k_group<<<grid1((long long)Bb * S3 * NS3), 256, 0, stream>>>(
      nxyz2, nxyz3, l2f, 256, gidx3, Abuf, S2, S3, NS3, pad32(259), (long long)Bb * S3 * NS3);
  const int M3 = Bb * S3 * NS3;  // 8192
  mlp_layer(26, M3, 259, 256, 1, nullptr, Abuf);
  mlp_layer(30, M3, 256, 512, 1, nullptr, Abuf);
  mlp_layer(34, M3, 512, 1024, 1, nullptr, Abuf);
  k_maxpool<<<grid1((long long)Bb * S3 * 1024), 256, 0, stream>>>(Abuf, l3f, NS3, 1024, (long long)Bb * S3 * 1024);

  // concat l3 with broadcast ecg -> feat (B, 16, 1152)
  k_concat_feat<<<grid1((long long)Bb * S3 * 1152), 256, 0, stream>>>(
      l3f, ecg, featf, S3, 1024, 128, (long long)Bb * S3 * 1152);

  // ---- FP3: (l2_xyz <- l3_xyz), concat l2 ----
  k_interp_concat<<<grid1((long long)Bb * S2), 256, 0, stream>>>(
      nxyz2, nxyz3, l2f, 256, featf, 1152, Abuf, S2, S3, 1408, (long long)Bb * S2);
  mlp_layer(38, Bb * S2, 1408, 256, 1, nullptr, Abuf);
  mlp_layer(42, Bb * S2, 256, 256, 1, l2n, (u16*)nullptr);

  // ---- FP2 ----
  k_interp_concat<<<grid1((long long)Bb * S1), 256, 0, stream>>>(
      nxyz1, nxyz2, l1f, 128, l2n, 256, Abuf, S1, S2, 384, (long long)Bb * S1);
  mlp_layer(46, Bb * S1, 384, 256, 1, nullptr, Abuf);
  mlp_layer(50, Bb * S1, 256, 128, 1, l1n, (u16*)nullptr);

  // ---- FP1 (points1 = None) ----
  k_interp_concat<<<grid1((long long)Bb * N0), 256, 0, stream>>>(
      xyz, nxyz1, (const float*)nullptr, 0, l1n, 128, Abuf, N0, S1, 128, (long long)Bb * N0);
  const int M0 = Bb * N0;  // 131072
  mlp_layer(54, M0, 128, 128, 1, nullptr, Abuf);
  mlp_layer(58, M0, 128, 128, 1, nullptr, Abuf);
  mlp_layer(62, M0, 128, 128, 1, nullptr, Abuf);

  // ---- head: linear 128->128 + BN + relu, then linear 128->10 into d_out ----
  mlp_layer(66, M0, 128, 128, 1, nullptr, Abuf);
  conv_wt(70, 128, 10, 128);
  gemm(Abuf, Wbf, F(71), nullptr, (float*)d_out, M0, 10, 128);
}
